// Block_3616362463321
// MI455X (gfx1250) — compile-verified, matrix operand order checked by reading, not run on hardware
//
#include <hip/hip_runtime.h>
#include <hip/hip_bf16.h>

// ---------------------------------------------------------------------------
// Types for CDNA5 WMMA (wave32, 16x16x32 bf16 -> f32)
// ---------------------------------------------------------------------------
typedef __attribute__((ext_vector_type(16))) __bf16 v16bf;
typedef __attribute__((ext_vector_type(8)))  float  v8f;
typedef __attribute__((ext_vector_type(8)))  unsigned int v8u;
typedef __attribute__((ext_vector_type(4)))  unsigned int v4u;

__device__ __forceinline__ unsigned short f2bf(float f) {
  __bf16 b = (__bf16)f;
  return __builtin_bit_cast(unsigned short, b);
}
__device__ __forceinline__ v16bf frag_cast(v8u r) {
  return __builtin_bit_cast(v16bf, r);
}
__device__ __forceinline__ v8f wmma_bf16(v8u a, v8u b, v8f c) {
  return __builtin_amdgcn_wmma_f32_16x16x32_bf16(
      false, frag_cast(a), false, frag_cast(b), (short)0, c, false, false);
}

// ISA 7.12.2 16-bit A/B fragment K pattern for lane half `hi`:
//   dword i<4 : k = 8*hi + 2*i ; i>=4 : k = 16 + 8*hi + 2*(i-4)
__device__ __forceinline__ int kpat(int i, int hi) {
  return (i < 4) ? (8 * hi + 2 * i) : (16 + 8 * hi + 2 * (i - 4));
}

// ---------------------------------------------------------------------------
// CDNA5 async global->LDS copy (ASYNCcnt-tracked, no VGPR round trip).
// Issued as inline asm: VDST = per-lane LDS byte offset (low 32 bits of the
// generic shared address per ISA ch.10 aperture rules), VADDR = 64-bit global
// address, off = no SADDR.
// ---------------------------------------------------------------------------
#if __has_builtin(__builtin_amdgcn_global_load_async_to_lds_b128)
#define HAVE_ASYNC_CP 1
#else
#define HAVE_ASYNC_CP 0
#warning "async-to-LDS not available on this toolchain: synchronous staging fallback"
#endif

__device__ __forceinline__ void cp_async_b128(void* lds, const void* g) {
#if HAVE_ASYNC_CP
  unsigned int lds_off = (unsigned int)(size_t)lds;
  asm volatile("global_load_async_to_lds_b128 %0, %1, off"
               :: "v"(lds_off), "v"(g)
               : "memory");
#else
  *(uint4*)lds = *(const uint4*)g;
#endif
}

template <int N>
__device__ __forceinline__ void wait_async() {
#if HAVE_ASYNC_CP
  asm volatile("s_wait_asynccnt %0" ::"n"(N) : "memory");
#endif
}

template <int N>
__device__ __forceinline__ void wait_tensor() {
#if __has_builtin(__builtin_amdgcn_s_wait_tensorcnt)
  __builtin_amdgcn_s_wait_tensorcnt((unsigned short)N);
#else
  asm volatile("s_wait_tensorcnt %0" ::"n"(N) : "memory");
#endif
}

// ---------------------------------------------------------------------------
// Tensor Data Mover: 2D tile load, global -> LDS.
// D# per ISA 08_async_tensor §8.3/8.4; VADDR2/3 omitted (NULL) => 2D form.
// group0: [1:0]=count=1, [63:32]=lds byte addr, [120:64]=global byte addr,
//         [127:126]=type=2.
// group1: [17:16]=data_size(1=2B), [79:48]=tensor_dim0, [111:80]=tensor_dim1,
//         [127:112]=tile_dim0, [143:128]=tile_dim1, [207:160]=dim0 stride.
// All operands wave-uniform; "s" constraints put D# in SGPR quad/octet.
// ---------------------------------------------------------------------------
__device__ __forceinline__ void tdm_load_2d(unsigned int lds_off, const void* g,
                                            unsigned int tile_d0,
                                            unsigned int tile_d1,
                                            unsigned int tensor_d0,
                                            unsigned int tensor_d1,
                                            unsigned int stride_d0) {
  unsigned long long ga = (unsigned long long)(size_t)g;
  v4u g0;
  g0.x = 1u;                                        // count=1, user desc
  g0.y = lds_off;                                   // LDS byte address
  g0.z = (unsigned int)ga;                          // global addr [31:0]
  g0.w = (unsigned int)(ga >> 32) | (2u << 30);     // addr[56:32] | type=2
  v8u g1;
  g1[0] = (1u << 16);                               // data_size = 2 bytes
  g1[1] = (tensor_d0 & 0xffffu) << 16;              // tensor_dim0[15:0]
  g1[2] = (tensor_d0 >> 16) | ((tensor_d1 & 0xffffu) << 16);
  g1[3] = (tensor_d1 >> 16) | (tile_d0 << 16);      // tile_dim0
  g1[4] = tile_d1 & 0xffffu;                        // tile_dim1 (tile_dim2=0)
  g1[5] = stride_d0;                                // dim0 stride [31:0]
  g1[6] = 0u;                                       // stride hi, dim1 stride lo
  g1[7] = 0u;
  asm volatile("tensor_load_to_lds %0, %1" :: "s"(g0), "s"(g1) : "memory");
}

// ---------------------------------------------------------------------------
// Weight packing: everything lands as bf16 [N][K] (K-contiguous rows) so the
// GEMM stages both operands with plain block copies (no in-kernel transpose).
// ---------------------------------------------------------------------------
__global__ void pack_qkv_w(const float* __restrict__ Wq,
                           const float* __restrict__ Wk,
                           const float* __restrict__ Wv,
                           unsigned short* __restrict__ out) {
  int idx = blockIdx.x * blockDim.x + threadIdx.x;   // [0, 3*16*1024*64)
  if (idx >= 3 * 1024 * 1024) return;
  int which = idx / (1024 * 1024);
  int rem   = idx % (1024 * 1024);
  int h  = rem / (1024 * 64);
  int r2 = rem % (1024 * 64);
  int c = r2 / 64, d = r2 % 64;
  const float* W = (which == 0) ? Wq : ((which == 1) ? Wk : Wv);
  float v = W[(h * 1024 + c) * 64 + d];
  // n = which*1024 + h*64 + d ; k = c
  out[(size_t)(which * 1024 + h * 64 + d) * 1024 + c] = f2bf(v);
}

__global__ void pack_w_t(const float* __restrict__ W,   // [K=1024][N=1024]
                         unsigned short* __restrict__ out) {  // [N][K]
  int idx = blockIdx.x * blockDim.x + threadIdx.x;
  if (idx >= 1024 * 1024) return;
  int k = idx >> 10, n = idx & 1023;                 // coalesced reads
  out[(size_t)n * 1024 + k] = f2bf(W[idx]);
}

// ---------------------------------------------------------------------------
// LayerNorm (f32 in, bf16 out). One block per row of C=1024.
// ---------------------------------------------------------------------------
__global__ __launch_bounds__(256) void ln_fwd_bf16(
    const float* __restrict__ x, const float* __restrict__ g,
    const float* __restrict__ be, unsigned short* __restrict__ out) {
  __shared__ float red0[8];
  __shared__ float red1[8];
  const int row = blockIdx.x, tid = threadIdx.x;
  const float4 xv = *(const float4*)(x + (size_t)row * 1024 + tid * 4);
  float xa[4] = {xv.x, xv.y, xv.z, xv.w};
  float s = 0.f, s2 = 0.f;
#pragma unroll
  for (int j = 0; j < 4; ++j) { s += xa[j]; s2 += xa[j] * xa[j]; }
#pragma unroll
  for (int d = 1; d < 32; d <<= 1) {
    s  += __shfl_xor(s,  d, 32);
    s2 += __shfl_xor(s2, d, 32);
  }
  if ((tid & 31) == 0) { red0[tid >> 5] = s; red1[tid >> 5] = s2; }
  __syncthreads();
  float ts = 0.f, ts2 = 0.f;
#pragma unroll
  for (int i = 0; i < 8; ++i) { ts += red0[i]; ts2 += red1[i]; }
  const float mean = ts * (1.0f / 1024.0f);
  const float var  = ts2 * (1.0f / 1024.0f) - mean * mean;
  const float rstd = rsqrtf(var + 1e-5f);
  unsigned int pack[2];
#pragma unroll
  for (int j = 0; j < 2; ++j) {
    int c0 = tid * 4 + 2 * j;
    float y0 = (xa[2 * j]     - mean) * rstd * g[c0]     + be[c0];
    float y1 = (xa[2 * j + 1] - mean) * rstd * g[c0 + 1] + be[c0 + 1];
    pack[j] = (unsigned int)f2bf(y0) | ((unsigned int)f2bf(y1) << 16);
  }
  *(uint2*)(out + (size_t)row * 1024 + tid * 4) = make_uint2(pack[0], pack[1]);
}

// ---------------------------------------------------------------------------
// Tiled bf16 WMMA GEMM: D[M,N] = A[M,K] * Bt[N,K]^T (+bias)(+res)(relu)
// Block tile 128x128, 8 waves as 2(M)x4(N), wave tile 64x32 = 4x2 WMMA,
// K-step 32, 2-stage LDS double buffering.
// A tile: per-lane async-to-LDS b128 (activation rows).
// B tile: one TDM tensor_load_to_lds per stage issued by wave 0 (weights).
// ---------------------------------------------------------------------------
constexpr int EPI_BF16_PLAIN     = 0;
constexpr int EPI_F32_BIAS_RES   = 1;
constexpr int EPI_BF16_BIAS_RELU = 2;

template <int MODE>
__global__ __launch_bounds__(256) void gemm_bf16_wmma(
    const unsigned short* __restrict__ A,     // [M,K] bf16
    const unsigned short* __restrict__ Bt,    // [N,K] bf16 (pre-transposed)
    const float* __restrict__ bias,           // [N] or null
    const float* __restrict__ Res,            // [M,N] f32 or null
    void* __restrict__ Dv,                    // f32 or bf16, ld = N
    int M, int N, int K) {
  __shared__ unsigned short sTile[2][2][128 * 32];   // [stage][A|B][row*32+k]

  const int tid = threadIdx.x;
  const int lane = tid & 31, w = tid >> 5;
  const int lr = lane & 15, hi = lane >> 4;
  const int m0 = blockIdx.y * 128, n0 = blockIdx.x * 128;
  const int wm = (w >> 2) * 64, wn = (w & 3) * 32;

  int kp[8];
#pragma unroll
  for (int i = 0; i < 8; ++i) kp[i] = kpat(i, hi);

  v8f acc[4][2] = {};
  const int nk = K >> 5;

  auto issue = [&](int s) {
    unsigned short* dA = sTile[s & 1][0];
    const int k0 = s * 32;
#pragma unroll
    for (int i = 0; i < 2; ++i) {
      int e = tid + i * 256;                 // [0,512) b128 chunks
      int row = e >> 2, q = (e & 3) * 8;     // 4 chunks per 32-elem row
      cp_async_b128(dA + row * 32 + q, A + (size_t)(m0 + row) * K + k0 + q);
    }
    if (w == 0) {                            // wave-uniform: TDM for B tile
      tdm_load_2d((unsigned int)(size_t)&sTile[s & 1][1][0],
                  Bt + (size_t)n0 * K + k0,
                  /*tile*/ 32, 128, /*tensor*/ (unsigned)K, (unsigned)N,
                  /*stride*/ (unsigned)K);
    }
  };

  issue(0);
  for (int s = 0; s < nk; ++s) {
    if (s + 1 < nk) {
      issue(s + 1);
      wait_async<2>();                       // stage s A-chunks landed
      if (w == 0) wait_tensor<1>();          // stage s B tile landed
    } else {
      wait_async<0>();
      if (w == 0) wait_tensor<0>();
    }
    __syncthreads();

    const unsigned int* lA = (const unsigned int*)sTile[s & 1][0];
    const unsigned int* lB = (const unsigned int*)sTile[s & 1][1];
    v8u ar[4], br[2];
#pragma unroll
    for (int mi = 0; mi < 4; ++mi) {
      int m = wm + mi * 16 + lr;
#pragma unroll
      for (int i = 0; i < 8; ++i) ar[mi][i] = lA[m * 16 + (kp[i] >> 1)];
    }
#pragma unroll
    for (int ni = 0; ni < 2; ++ni) {
      int n = wn + ni * 16 + lr;
#pragma unroll
      for (int i = 0; i < 8; ++i) br[ni][i] = lB[n * 16 + (kp[i] >> 1)];
    }
#pragma unroll
    for (int mi = 0; mi < 4; ++mi)
#pragma unroll
      for (int ni = 0; ni < 2; ++ni)
        acc[mi][ni] = wmma_bf16(ar[mi], br[ni], acc[mi][ni]);
    __syncthreads();   // protect stage being refilled next iteration
  }

  // Epilogue: C/D layout — lane lr = column, VGPR r -> row r + 8*hi.
#pragma unroll
  for (int mi = 0; mi < 4; ++mi)
#pragma unroll
    for (int ni = 0; ni < 2; ++ni) {
      int col = n0 + wn + ni * 16 + lr;
      float bv = (MODE != EPI_BF16_PLAIN) ? bias[col] : 0.f;
#pragma unroll
      for (int r = 0; r < 8; ++r) {
        int row = m0 + wm + mi * 16 + r + 8 * hi;
        float v = acc[mi][ni][r] + bv;
        size_t off = (size_t)row * N + col;
        if (MODE == EPI_F32_BIAS_RES) {
          ((float*)Dv)[off] = v + Res[off];
        } else if (MODE == EPI_BF16_BIAS_RELU) {
          ((unsigned short*)Dv)[off] = f2bf(fmaxf(v, 0.f));
        } else {
          ((unsigned short*)Dv)[off] = f2bf(v);
        }
      }
    }
}

// ---------------------------------------------------------------------------
// Flash attention (causal). qkv: [B*T, 3072] bf16 (q|k|v each [*,1024]).
// Block = 4 waves; wave owns 16 query rows; key blocks of 32.
// V tile staged per-wave by ONE TDM op per key block (overlapped with the
// S WMMAs + softmax); P bounced C-layout -> A-layout through per-wave LDS.
// All sync is wave-local (s_wait_tensorcnt / s_wait_dscnt), so divergent
// causal trip counts across waves need no barriers.
// ---------------------------------------------------------------------------
__global__ __launch_bounds__(128) void attn_flash(
    const unsigned short* __restrict__ qkv,
    unsigned short* __restrict__ attnOut) {
  __shared__ unsigned short ldsP[4][16 * 32];
  __shared__ unsigned short sV[4][32 * 64];
  const int T = 1024, C3 = 3072;
  const int tid = threadIdx.x;
  const int lane = tid & 31, w = tid >> 5;
  const int lr = lane & 15, hi = lane >> 4;
  const int bh = blockIdx.y;
  const int b = bh >> 4, h = bh & 15;
  const int qt0 = blockIdx.x * 64 + w * 16;
  const size_t rowb = (size_t)b * T;
  const float scale = 0.125f;   // Dh^-0.5, Dh=64

  int kp[8];
#pragma unroll
  for (int i = 0; i < 8; ++i) kp[i] = kpat(i, hi);

  // Q fragments: A layout, lane -> row m=lr, two K-halves of Dh=64.
  v8u qf[2];
#pragma unroll
  for (int ci = 0; ci < 2; ++ci)
#pragma unroll
    for (int i = 0; i < 8; ++i)
      qf[ci][i] = *(const unsigned int*)(
          qkv + (rowb + qt0 + lr) * (size_t)C3 + h * 64 + ci * 32 + kp[i]);

  v8f O[4] = {};
  float Mr[8], Lr[8];
#pragma unroll
  for (int r = 0; r < 8; ++r) { Mr[r] = -1e30f; Lr[r] = 0.f; }

  for (int kb = 0; kb <= qt0 + 15; kb += 32) {
    // ---- one TDM op stages this key block's V tile (32 keys x 64 dims) ----
    tdm_load_2d((unsigned int)(size_t)&sV[w][0],
                qkv + (rowb + kb) * (size_t)C3 + 2048 + h * 64,
                /*tile*/ 64, 32, /*tensor*/ (unsigned)C3, (unsigned)(8 * T),
                /*stride*/ (unsigned)C3);

    // ---- scores: two 16x16 f32 tiles over 32 keys ----
    v8f s0 = {}, s1 = {};
#pragma unroll
    for (int ci = 0; ci < 2; ++ci) {
      v8u kf0, kf1;
#pragma unroll
      for (int i = 0; i < 8; ++i) {
        kf0[i] = *(const unsigned int*)(
            qkv + (rowb + kb + lr) * (size_t)C3 + 1024 + h * 64 + ci * 32 + kp[i]);
        kf1[i] = *(const unsigned int*)(
            qkv + (rowb + kb + 16 + lr) * (size_t)C3 + 1024 + h * 64 + ci * 32 + kp[i]);
      }
      s0 = wmma_bf16(qf[ci], kf0, s0);
      s1 = wmma_bf16(qf[ci], kf1, s1);
    }

    // ---- online softmax: per-register row, reduce across 16 columns ----
#pragma unroll
    for (int r = 0; r < 8; ++r) {
      const int m = r + 8 * hi;
      const int t = qt0 + m;
      float v0 = s0[r] * scale;
      float v1 = s1[r] * scale;
      if (kb + lr > t)      v0 = -1e30f;   // causal mask
      if (kb + 16 + lr > t) v1 = -1e30f;
      float rm = fmaxf(v0, v1);
#pragma unroll
      for (int d = 1; d < 16; d <<= 1) rm = fmaxf(rm, __shfl_xor(rm, d, 32));
      const float Mn = fmaxf(Mr[r], rm);
      const float al = __expf(Mr[r] - Mn);
      const float p0 = __expf(v0 - Mn);
      const float p1 = __expf(v1 - Mn);
      float rs = p0 + p1;
#pragma unroll
      for (int d = 1; d < 16; d <<= 1) rs += __shfl_xor(rs, d, 32);
      Lr[r] = Lr[r] * al + rs;
      Mr[r] = Mn;
#pragma unroll
      for (int f = 0; f < 4; ++f) O[f][r] *= al;
      ldsP[w][m * 32 + lr]      = f2bf(p0);
      ldsP[w][m * 32 + 16 + lr] = f2bf(p1);
    }
    // LDS in-order within wave; wait for P stores before transposed reads.
    asm volatile("s_wait_dscnt 0" ::: "memory");

    // ---- P: re-load in A-fragment layout (16 rows x 32 keys) ----
    v8u pf;
    const unsigned int* lp = (const unsigned int*)ldsP[w];
#pragma unroll
    for (int i = 0; i < 8; ++i) pf[i] = lp[lr * 16 + (kp[i] >> 1)];

    // ---- V tile landed? then O += P * V over 4 d-tiles of 16 ----
    wait_tensor<0>();
#pragma unroll
    for (int di = 0; di < 4; ++di) {
      v8u vf;
      const unsigned short* vb = &sV[w][di * 16 + lr];
#pragma unroll
      for (int i = 0; i < 8; ++i) {
        int k = kp[i];
        unsigned int lo = vb[k * 64];
        unsigned int hh = vb[(k + 1) * 64];
        vf[i] = lo | (hh << 16);
      }
      O[di] = wmma_bf16(pf, vf, O[di]);
    }
  }

  // ---- normalize and store (concat-heads layout [B*T, C]) ----
#pragma unroll
  for (int di = 0; di < 4; ++di)
#pragma unroll
    for (int r = 0; r < 8; ++r) {
      float ov = O[di][r] / Lr[r];
      attnOut[(rowb + qt0 + r + 8 * hi) * 1024ull + h * 64 + di * 16 + lr] =
          f2bf(ov);
    }
}

// ---------------------------------------------------------------------------
// Orchestration
// ---------------------------------------------------------------------------
extern "C" void kernel_launch(void* const* d_in, const int* in_sizes, int n_in,
                              void* d_out, int out_size, void* d_ws, size_t ws_size,
                              hipStream_t stream) {
  const float* x     = (const float*)d_in[0];
  const float* Wq    = (const float*)d_in[1];
  const float* Wk    = (const float*)d_in[2];
  const float* Wv    = (const float*)d_in[3];
  const float* Wproj = (const float*)d_in[4];
  const float* bproj = (const float*)d_in[5];
  const float* W1    = (const float*)d_in[6];
  const float* b1    = (const float*)d_in[7];
  const float* W2    = (const float*)d_in[8];
  const float* b2    = (const float*)d_in[9];
  const float* g1    = (const float*)d_in[10];
  const float* be1   = (const float*)d_in[11];
  const float* g2    = (const float*)d_in[12];
  const float* be2   = (const float*)d_in[13];

  char* w = (char*)d_ws;
  const size_t MB = 1ull << 20;
  unsigned short* xn    = (unsigned short*)(w);              // 16 MB
  unsigned short* wqkv  = (unsigned short*)(w + 16 * MB);    //  6 MB [3072][1024]
  unsigned short* wpb   = (unsigned short*)(w + 22 * MB);    //  2 MB [N][K]
  unsigned short* w1b   = (unsigned short*)(w + 24 * MB);    //  2 MB [N][K]
  unsigned short* w2b   = (unsigned short*)(w + 26 * MB);    //  2 MB [N][K]
  unsigned short* qkv   = (unsigned short*)(w + 28 * MB);    // 48 MB
  unsigned short* attnb = (unsigned short*)(w + 76 * MB);    // 16 MB
  float*          x1    = (float*)(w + 92 * MB);             // 32 MB
  unsigned short* xn2   = (unsigned short*)(w + 124 * MB);   // 16 MB
  unsigned short* hb    = (unsigned short*)(w + 140 * MB);   // 16 MB

  // Weight packing (amortized; all outputs bf16 [N][K]).
  pack_qkv_w<<<(3 * 1024 * 1024 + 255) / 256, 256, 0, stream>>>(Wq, Wk, Wv, wqkv);
  pack_w_t<<<4096, 256, 0, stream>>>(Wproj, wpb);
  pack_w_t<<<4096, 256, 0, stream>>>(W1, w1b);
  pack_w_t<<<4096, 256, 0, stream>>>(W2, w2b);

  // xn = LN(x); fused bf16 downconvert
  ln_fwd_bf16<<<8192, 256, 0, stream>>>(x, g1, be1, xn);

  // qkv = xn @ Wqkv   [8192 x 3072]
  gemm_bf16_wmma<EPI_BF16_PLAIN><<<dim3(24, 64), 256, 0, stream>>>(
      xn, wqkv, nullptr, nullptr, qkv, 8192, 3072, 1024);

  // causal flash attention -> attnb [8192 x 1024]
  attn_flash<<<dim3(16, 128), 128, 0, stream>>>(qkv, attnb);

  // x1 = x + attnb @ Wproj + bproj
  gemm_bf16_wmma<EPI_F32_BIAS_RES><<<dim3(8, 64), 256, 0, stream>>>(
      attnb, wpb, bproj, x, x1, 8192, 1024, 1024);

  // xn2 = LN(x1)
  ln_fwd_bf16<<<8192, 256, 0, stream>>>(x1, g2, be2, xn2);

  // hb = relu(xn2 @ W1 + b1)
  gemm_bf16_wmma<EPI_BF16_BIAS_RELU><<<dim3(8, 64), 256, 0, stream>>>(
      xn2, w1b, b1, nullptr, hb, 8192, 1024, 1024);

  // out = x1 + hb @ W2 + b2
  gemm_bf16_wmma<EPI_F32_BIAS_RES><<<dim3(8, 64), 256, 0, stream>>>(
      hb, w2b, b2, x1, (float*)d_out, 8192, 1024, 1024);
}
